// MultiHead_50096498541092
// MI455X (gfx1250) — compile-verified, matrix operand order checked by reading, not run on hardware
//
#include <hip/hip_runtime.h>
#include <hip/hip_bf16.h>

// Multi-head causal attention fwd, bf16 WMMA everywhere, flash-style softmax.
// B=2 T=2048 C=1024 H=16 D=64.  scale = C^-0.5 = 2^-5 exactly -> folded into Q.
// Round 2: software-pipelined fragment loads, LDS-transposed W conversion,
//          V loads hidden under softmax, global_prefetch for next s-block.

#define B_ 2
#define T_ 2048
#define C_ 1024
#define H_ 16
#define D_ 64

typedef __attribute__((ext_vector_type(16))) __bf16 bf16x16;
typedef __attribute__((ext_vector_type(8)))  float  f32x8;

union AF { unsigned int u[8]; bf16x16 v; };
union CF { float f[8]; f32x8 v; };

__device__ __forceinline__ unsigned short f2bf(float f) {
    unsigned int u = __float_as_uint(f);
    u += 0x7FFFu + ((u >> 16) & 1u);          // round-to-nearest-even
    return (unsigned short)(u >> 16);
}

// K-offset of packed pair j (0..7) for the 16-bit A/B fragment layout:
// VGPR j<4 : K = half*8 + j*2 ; VGPR j>=4 : K = 16 + half*8 + (j-4)*2
__device__ __forceinline__ int koff(int j, int half) {
    return ((j < 4) ? 0 : 16) + half * 8 + (j & 3) * 2;
}

// load one 16x32 bf16 A/B fragment whose K dimension is contiguous at `src`
__device__ __forceinline__ void load_frag(AF& a, const unsigned short* src, int half) {
    #pragma unroll
    for (int j = 0; j < 8; ++j)
        a.u[j] = *(const unsigned int*)(src + koff(j, half));
}

__device__ __forceinline__ f32x8 wmma_bf16(const AF& a, const AF& b, f32x8 c) {
    return __builtin_amdgcn_wmma_f32_16x16x32_bf16(
        false, a.v, false, b.v, (short)0, c, false, false);
}

// ---------------------------------------------------------------- converters
__global__ void cvt_x_kernel(const float* __restrict__ x,
                             unsigned short* __restrict__ xb, int n) {
    int i = blockIdx.x * blockDim.x + threadIdx.x;
    if (i < n) xb[i] = f2bf(x[i]);
}

// Wt[p][h][d][c] = W_p[h][c][d]  via LDS tile transpose (both sides coalesced).
// One block handles a 64(c) x 64(d) tile: grid = 3 * H * (C/64) = 768 blocks.
__global__ __launch_bounds__(256)
void cvt_w_kernel(const float* __restrict__ wq,
                  const float* __restrict__ wk,
                  const float* __restrict__ wv,
                  unsigned short* __restrict__ wt) {
    __shared__ unsigned short tile[64][65];   // +1 pad: no bank conflicts
    int bid = blockIdx.x;
    const int cblk = bid & 15;  bid >>= 4;    // C/64 = 16
    const int h    = bid & 15;  bid >>= 4;
    const int p    = bid;                     // 0..2
    const int c0   = cblk * 64;
    const float* w = (p == 0) ? wq : ((p == 1) ? wk : wv);

    #pragma unroll
    for (int i = threadIdx.x; i < 4096; i += 256) {
        const int d = i & 63, c = i >> 6;     // consecutive threads -> consecutive d
        tile[c][d] = f2bf(w[((size_t)h * C_ + c0 + c) * D_ + d]);
    }
    __syncthreads();
    #pragma unroll
    for (int i = threadIdx.x; i < 4096; i += 256) {
        const int c = i & 63, d = i >> 6;     // consecutive threads -> consecutive c
        wt[((size_t)((p * H_ + h) * D_) + d) * C_ + c0 + c] = tile[c][d];
    }
}

// ---------------------------------------------------------------- projection
// One wave computes a 16(t) x 64(d) strip of Q, K or V for one (b,h).
// Q,K stored [B,H,T,D] bf16 ; V stored transposed [B,H,D,T] bf16.
// k-loop is double-buffered: fragments for step n+1 load while step n computes.
__global__ __launch_bounds__(256)
void proj_kernel(const unsigned short* __restrict__ xb,
                 const unsigned short* __restrict__ wt,
                 unsigned short* __restrict__ qb,
                 unsigned short* __restrict__ kb,
                 unsigned short* __restrict__ vtb) {
    const int lane = threadIdx.x & 31;
    const int wave = threadIdx.x >> 5;
    const int half = lane >> 4;
    const int ln16 = lane & 15;

    int gw = blockIdx.x * 8 + wave;
    const int ttile = gw & 127;  gw >>= 7;    // T/16 = 128
    const int h     = gw & 15;   gw >>= 4;
    const int b     = gw & 1;
    const int p     = gw >> 1;                // 0=Q 1=K 2=V
    const int t0    = ttile * 16;

    const unsigned short* arow = xb + (size_t)(b * T_ + t0 + ln16) * C_;
    const unsigned short* brow = wt + ((size_t)((p * H_ + h) * D_) + ln16) * C_;

    CF acc[4];
    #pragma unroll
    for (int dt = 0; dt < 4; ++dt)
        #pragma unroll
        for (int r = 0; r < 8; ++r) acc[dt].f[r] = 0.f;

    AF a0, a1, b0[4], b1[4];
    load_frag(a0, arow, half);
    #pragma unroll
    for (int dt = 0; dt < 4; ++dt)
        load_frag(b0[dt], brow + (size_t)(dt * 16) * C_, half);

    for (int k0 = 0; k0 < C_; k0 += 64) {
        const int kn1 = k0 + 32;
        const int kn2 = (k0 + 64) & (C_ - 1);   // wraps to 0 on last iter (unused)

        load_frag(a1, arow + kn1, half);
        #pragma unroll
        for (int dt = 0; dt < 4; ++dt)
            load_frag(b1[dt], brow + (size_t)(dt * 16) * C_ + kn1, half);
        #pragma unroll
        for (int dt = 0; dt < 4; ++dt)
            acc[dt].v = wmma_bf16(a0, b0[dt], acc[dt].v);

        load_frag(a0, arow + kn2, half);
        #pragma unroll
        for (int dt = 0; dt < 4; ++dt)
            load_frag(b0[dt], brow + (size_t)(dt * 16) * C_ + kn2, half);
        #pragma unroll
        for (int dt = 0; dt < 4; ++dt)
            acc[dt].v = wmma_bf16(a1, b1[dt], acc[dt].v);
    }

    const float scale = (p == 0) ? 0.03125f : 1.0f;   // C^-0.5 folded into Q
    if (p < 2) {
        unsigned short* dst = (p == 0 ? qb : kb) + (size_t)((b * H_ + h) * T_) * D_;
        #pragma unroll
        for (int dt = 0; dt < 4; ++dt)
            #pragma unroll
            for (int r = 0; r < 8; ++r)
                dst[(size_t)(t0 + r + 8 * half) * D_ + dt * 16 + ln16] =
                    f2bf(acc[dt].f[r] * scale);
    } else {
        unsigned short* dst = vtb + (size_t)((b * H_ + h) * D_) * T_;
        #pragma unroll
        for (int dt = 0; dt < 4; ++dt)
            #pragma unroll
            for (int r = 0; r < 8; ++r)
                dst[(size_t)(dt * 16 + ln16) * T_ + t0 + r + 8 * half] =
                    f2bf(acc[dt].f[r]);
    }
}

// ---------------------------------------------------------------- attention
// Block = 8 waves; each wave owns 16 query rows; flash online softmax.
__global__ __launch_bounds__(256)
void attn_kernel(const unsigned short* __restrict__ qb,
                 const unsigned short* __restrict__ kb,
                 const unsigned short* __restrict__ vtb,
                 float* __restrict__ out) {
    __shared__ __align__(16) unsigned short pbuf[8][16 * 64];   // bf16 P per wave

    const int lane = threadIdx.x & 31;
    const int wave = threadIdx.x >> 5;
    const int half = lane >> 4;
    const int ln16 = lane & 15;

    int bidx = blockIdx.x;
    const int qblk = bidx & 15;  bidx >>= 4;   // T/128 = 16
    const int h    = bidx & 15;  bidx >>= 4;
    const int b    = bidx;

    const int t0 = qblk * 128 + wave * 16;

    const unsigned short* qrow  = qb  + ((size_t)((b * H_ + h) * T_) + t0 + ln16) * D_;
    const unsigned short* kbase = kb  + (size_t)((b * H_ + h) * T_) * D_;
    const unsigned short* vbase = vtb + (size_t)((b * H_ + h) * D_) * T_;

    // Q fragments (16 x 64 split as two 16x32 A-frags), loaded once
    AF qa0, qa1;
    load_frag(qa0, qrow, half);
    load_frag(qa1, qrow + 32, half);

    CF o[4];
    #pragma unroll
    for (int dt = 0; dt < 4; ++dt)
        #pragma unroll
        for (int r = 0; r < 8; ++r) o[dt].f[r] = 0.f;

    float m_i[8], l_i[8];
    #pragma unroll
    for (int r = 0; r < 8; ++r) { m_i[r] = -__builtin_inff(); l_i[r] = 0.f; }

    unsigned short* pw = pbuf[wave];
    const int tmax = t0 + 15;

    for (int s0 = 0; s0 <= tmax; s0 += 64) {
        // prefetch next s-block K/V rows into cache while this block computes
        if (s0 + 64 <= tmax) {
            __builtin_prefetch(kbase + (size_t)(s0 + 64 + ln16) * D_, 0, 3);
            __builtin_prefetch(vbase + (size_t)ln16 * T_ + s0 + 64, 0, 3);
        }

        // ---- scores: 16 x 64 tile; K fragments double-buffered across n-tiles
        CF sacc[4];
        #pragma unroll
        for (int st = 0; st < 4; ++st)
            #pragma unroll
            for (int r = 0; r < 8; ++r) sacc[st].f[r] = 0.f;

        AF kf[2][2];
        {
            const unsigned short* krow = kbase + (size_t)(s0 + ln16) * D_;
            load_frag(kf[0][0], krow, half);
            load_frag(kf[0][1], krow + 32, half);
        }
        #pragma unroll
        for (int st = 0; st < 4; ++st) {
            const int cur = st & 1, nxt = cur ^ 1;
            if (st < 3) {
                const unsigned short* krow =
                    kbase + (size_t)(s0 + (st + 1) * 16 + ln16) * D_;
                load_frag(kf[nxt][0], krow, half);
                load_frag(kf[nxt][1], krow + 32, half);
            }
            sacc[st].v = wmma_bf16(qa0, kf[cur][0], sacc[st].v);
            sacc[st].v = wmma_bf16(qa1, kf[cur][1], sacc[st].v);
        }

        // ---- issue V loads (s chunk 0) now; latency hides under softmax VALU
        AF vb0[4];
        #pragma unroll
        for (int dt = 0; dt < 4; ++dt)
            load_frag(vb0[dt], vbase + (size_t)(dt * 16 + ln16) * T_ + s0, half);

        // ---- causal mask (only needed near the diagonal)
        if (s0 + 63 > t0) {
            #pragma unroll
            for (int st = 0; st < 4; ++st) {
                const int s = s0 + st * 16 + ln16;
                #pragma unroll
                for (int r = 0; r < 8; ++r)
                    if (s > t0 + r + 8 * half) sacc[st].f[r] = -__builtin_inff();
            }
        }

        // ---- online softmax (rows live across 16 lanes of each half-wave)
        float fac[8];
        #pragma unroll
        for (int r = 0; r < 8; ++r) {
            float mx = fmaxf(fmaxf(sacc[0].f[r], sacc[1].f[r]),
                             fmaxf(sacc[2].f[r], sacc[3].f[r]));
            mx = fmaxf(mx, __shfl_xor(mx, 1, 16));
            mx = fmaxf(mx, __shfl_xor(mx, 2, 16));
            mx = fmaxf(mx, __shfl_xor(mx, 4, 16));
            mx = fmaxf(mx, __shfl_xor(mx, 8, 16));
            const float mnew = fmaxf(m_i[r], mx);
            const float f    = __expf(m_i[r] - mnew);
            m_i[r] = mnew;
            float sum = 0.f;
            #pragma unroll
            for (int st = 0; st < 4; ++st) {
                const float pv = __expf(sacc[st].f[r] - mnew);
                sacc[st].f[r] = pv;
                sum += pv;
            }
            sum += __shfl_xor(sum, 1, 16);
            sum += __shfl_xor(sum, 2, 16);
            sum += __shfl_xor(sum, 4, 16);
            sum += __shfl_xor(sum, 8, 16);
            l_i[r] = l_i[r] * f + sum;
            fac[r] = f;
        }

        // ---- P -> LDS (bf16) to relayout D-frag -> A-frag
        #pragma unroll
        for (int st = 0; st < 4; ++st)
            #pragma unroll
            for (int r = 0; r < 8; ++r)
                pw[(r + 8 * half) * 64 + st * 16 + ln16] = f2bf(sacc[st].f[r]);

        // rescale O while the DS stores are in flight
        #pragma unroll
        for (int dt = 0; dt < 4; ++dt)
            #pragma unroll
            for (int r = 0; r < 8; ++r) o[dt].f[r] *= fac[r];

        asm volatile("s_wait_dscnt 0" ::: "memory");

        // ---- O += P x V  (K = s in chunks of 32, V transposed so K contiguous)
        AF pa0, pa1, vb1[4];
        load_frag(pa0, pw + ln16 * 64, half);
        load_frag(pa1, pw + ln16 * 64 + 32, half);
        #pragma unroll
        for (int dt = 0; dt < 4; ++dt)   // s chunk 1 V loads overlap chunk-0 MMAs
            load_frag(vb1[dt], vbase + (size_t)(dt * 16 + ln16) * T_ + s0 + 32, half);
        #pragma unroll
        for (int dt = 0; dt < 4; ++dt)
            o[dt].v = wmma_bf16(pa0, vb0[dt], o[dt].v);
        #pragma unroll
        for (int dt = 0; dt < 4; ++dt)
            o[dt].v = wmma_bf16(pa1, vb1[dt], o[dt].v);
    }

    // ---- normalize and store [B,T,H*D] f32
    float inv[8];
    #pragma unroll
    for (int r = 0; r < 8; ++r) inv[r] = 1.0f / l_i[r];
    #pragma unroll
    for (int dt = 0; dt < 4; ++dt)
        #pragma unroll
        for (int r = 0; r < 8; ++r) {
            const int t = t0 + r + 8 * half;
            out[(size_t)(b * T_ + t) * (H_ * D_) + h * D_ + dt * 16 + ln16] =
                o[dt].f[r] * inv[r];
        }
}

// ---------------------------------------------------------------- launcher
extern "C" void kernel_launch(void* const* d_in, const int* in_sizes, int n_in,
                              void* d_out, int out_size, void* d_ws, size_t ws_size,
                              hipStream_t stream) {
    const float* x  = (const float*)d_in[0];
    const float* Wq = (const float*)d_in[1];
    const float* Wk = (const float*)d_in[2];
    const float* Wv = (const float*)d_in[3];
    float* out = (float*)d_out;

    // workspace carve (bytes):
    //   xb  bf16 [B,T,C]      @ 0         (8,388,608)
    //   Wt  bf16 [3,H,D,C]    @ 8388608   (6,291,456)
    //   Q   bf16 [B,H,T,D]    @ 14680064  (8,388,608)
    //   K   bf16 [B,H,T,D]    @ 23068672  (8,388,608)
    //   Vt  bf16 [B,H,D,T]    @ 31457280  (8,388,608)   total ~38 MB
    char* ws = (char*)d_ws;
    unsigned short* xb  = (unsigned short*)(ws);
    unsigned short* wt  = (unsigned short*)(ws + 8388608);
    unsigned short* qb  = (unsigned short*)(ws + 14680064);
    unsigned short* kb  = (unsigned short*)(ws + 23068672);
    unsigned short* vtb = (unsigned short*)(ws + 31457280);

    const int NX = B_ * T_ * C_;        // 4,194,304

    cvt_x_kernel<<<(NX + 255) / 256, 256, 0, stream>>>(x, xb, NX);
    cvt_w_kernel<<<3 * H_ * (C_ / 64), 256, 0, stream>>>(Wq, Wk, Wv, wt);

    // 3 proj * B * H * (T/16) waves = 12288 waves / 8 per block
    proj_kernel<<<1536, 256, 0, stream>>>(xb, wt, qb, kb, vtb);

    // B * H * (T/128) blocks
    attn_kernel<<<B_ * H_ * (T_ / 128), 256, 0, stream>>>(qb, kb, vtb, out);
}
